// PerformerBlock_57440892617547
// MI455X (gfx1250) — matrix operand order repaired and checked by reading, not compiled, not run on hardware
//
#include <hip/hip_runtime.h>
#include <math.h>

// ---------------------------------------------------------------------------
// Types
// ---------------------------------------------------------------------------
typedef __bf16 bf16;
typedef __attribute__((ext_vector_type(16))) __bf16 v16bf;
typedef __attribute__((ext_vector_type(8)))  float  v8f;

// ---------------------------------------------------------------------------
// Problem dimensions (Performer block)
// ---------------------------------------------------------------------------
constexpr int  Lnum = 6;
constexpr long Bb = 8, Nn = 2048, Dm = 512, Hh = 8, Dh = 64, Mm = 256;
constexpr long Rr  = Bb * Nn;    // 16384 token rows
constexpr long BH  = Bb * Hh;    // 64 batch-head pairs
constexpr long BHN = BH * Nn;    // 131072 (b,h,n) rows

// GEMM tiling: 128 threads (4 waves); block tile 128x64, K-step 32.
// Each wave computes 32 rows x 64 cols = 8 wmma tiles, reusing B fragments.
constexpr int BM = 128, BN = 64, BK = 32;
constexpr int APAD = 40;   // LDS row stride (bf16) for A tile (K-major rows)
constexpr int BPAD = 72;   // LDS row stride (bf16) for B tile (N-major rows)
constexpr int ASZ  = BM * APAD;       // elems per A buffer (10240 B)
constexpr int BSZ  = BK * BPAD;       // elems per B buffer (4608 B)

enum { G_RESID = 1, G_GELU = 2, G_SCALE = 4, G_BIAS = 8, G_F32 = 16 };

struct GemmP {
  const bf16* A;  const bf16* B;
  float* outF;    bf16* outB;
  const float* bias;   // [Ncols]
  const float* scale;  // per-row scale, scale[z*srow + row]
  long arow, brow, orow;              // row strides (elements)
  long abase, bbase, obase;           // constant base offsets (elements)
  long azdiv, azo, azi;               // a base += (z/azdiv)*azo + (z%azdiv)*azi
  long bzdiv, bzo, bzi;
  long ozdiv, ozo, ozi;
  long ordiv, oro;                    // out row split: (row/ordiv)*oro + (row%ordiv)*orow
  long srow;
  int  K;
};

// CDNA5 async DMA: global -> LDS, 16B per lane, tracked by ASYNCcnt.
__device__ __forceinline__ void async_ld_b128(unsigned lds_addr, const bf16* g) {
  asm volatile("global_load_async_to_lds_b128 %0, %1, off"
               :: "v"(lds_addr), "v"(g) : "memory");
}
__device__ __forceinline__ unsigned lds_off(const bf16* p) {
  return (unsigned)(size_t)p;   // low 32 bits of LDS flat address = LDS offset
}

// ---------------------------------------------------------------------------
// WMMA GEMM: C[M,N] = A[M,K] @ B[K,N]  (bf16 in, f32 accumulate)
// Double-buffered LDS tiles fed by global_load_async_to_lds_b128.
// A fragment: lane = M row, VGPR pairs = K (ISA 16-bit A layout).
// B fragment: lane = N col, packed elems = K (ISA 16-bit B layout).
// ---------------------------------------------------------------------------
template<int MODE>
__global__ __launch_bounds__(128)
void k_gemm_wmma(GemmP p) {
  __shared__ __align__(16) bf16 As[2][ASZ];
  __shared__ __align__(16) bf16 Bs[2][BSZ];

  const int tid  = threadIdx.x;
  const int wv   = tid >> 5;
  const int lane = tid & 31;
  const int hlf  = lane >> 4;
  const int r16  = lane & 15;
  const long z   = blockIdx.z;

  const long abase = p.abase + (z / p.azdiv) * p.azo + (z % p.azdiv) * p.azi;
  const long bbase = p.bbase + (z / p.bzdiv) * p.bzo + (z % p.bzdiv) * p.bzi;
  const long obase = p.obase + (z / p.ozdiv) * p.ozo + (z % p.ozdiv) * p.ozi;
  const long rowBase = (long)blockIdx.y * BM;
  const long colBase = (long)blockIdx.x * BN;

  v8f acc[2][4];
#pragma unroll
  for (int s = 0; s < 2; ++s)
#pragma unroll
    for (int t = 0; t < 4; ++t)
#pragma unroll
      for (int i = 0; i < 8; ++i) acc[s][t][i] = 0.f;

  // staging coordinates: A = 4 x 16B per thread (rows tid>>2 +0/32/64/96),
  //                      B = 2 x 16B per thread (rows tid>>3 +0/16)
  const int arr = tid >> 2, ac8 = (tid & 3) * 8;
  const int brr = tid >> 3, bc8 = (tid & 7) * 8;
  const bf16* gA = p.A + abase + (rowBase + arr) * p.arow + ac8;
  const bf16* gB = p.B + bbase + (long)brr * p.brow + colBase + bc8;
  const long a32 = 32 * p.arow;            // row stride chunks for A staging
  const long bK  = (long)BK * p.brow;      // advance B by 32 K-rows per tile
  const unsigned lA = lds_off(&As[0][arr * APAD + ac8]);
  const unsigned lB = lds_off(&Bs[0][brr * BPAD + bc8]);

  auto stage = [&](int buf) {
    const unsigned la = lA + (unsigned)buf * (ASZ * 2);
    async_ld_b128(la,                 gA);
    async_ld_b128(la + 32 * APAD * 2, gA + a32);
    async_ld_b128(la + 64 * APAD * 2, gA + 2 * a32);
    async_ld_b128(la + 96 * APAD * 2, gA + 3 * a32);
    const unsigned lb = lB + (unsigned)buf * (BSZ * 2);
    async_ld_b128(lb,                 gB);
    async_ld_b128(lb + 16 * BPAD * 2, gB + 16 * p.brow);
    gA += BK;
    gB += bK;
  };

  const int nk = p.K / BK;
  stage(0);
  for (int kb = 0; kb < nk; ++kb) {
    const int cur = kb & 1;
    if (kb + 1 < nk) {
      stage(cur ^ 1);                                   // overlap next tile DMA
      asm volatile("s_wait_asynccnt 0x6" ::: "memory"); // my tile-kb loads done
    } else {
      asm volatile("s_wait_asynccnt 0x0" ::: "memory");
    }
    __syncthreads();   // everyone's DMA for tile kb landed

    union Frag { int4 q[2]; v16bf v; } a0, a1, b;
    const bf16* a0s = &As[cur][(wv * 32 + r16) * APAD];
    const bf16* a1s = a0s + 16 * APAD;
    a0.q[0] = *(const int4*)(a0s + hlf * 8);        // K = hlf*8 + 0..7
    a0.q[1] = *(const int4*)(a0s + 16 + hlf * 8);   // K = 16 + hlf*8 + 0..7
    a1.q[0] = *(const int4*)(a1s + hlf * 8);
    a1.q[1] = *(const int4*)(a1s + 16 + hlf * 8);
#pragma unroll
    for (int nt = 0; nt < 4; ++nt) {
      const bf16* bs = &Bs[cur][nt * 16 + r16];  // N col = r16, K = hlf*8+s / 16+hlf*8+s
#pragma unroll
      for (int s = 0; s < 8; ++s) {
        b.v[s]     = bs[(hlf * 8 + s) * BPAD];
        b.v[8 + s] = bs[(16 + hlf * 8 + s) * BPAD];
      }
      acc[0][nt] = __builtin_amdgcn_wmma_f32_16x16x32_bf16(
          false, a0.v, false, b.v, (short)0, acc[0][nt], false, false);
      acc[1][nt] = __builtin_amdgcn_wmma_f32_16x16x32_bf16(
          false, a1.v, false, b.v, (short)0, acc[1][nt], false, false);
    }
    __syncthreads();   // done reading buf[cur] before it is re-filled
  }

  // Epilogue: C layout = lane(N), VGPR i -> M = hlf*8 + i
#pragma unroll
  for (int sub = 0; sub < 2; ++sub) {
#pragma unroll
    for (int nt = 0; nt < 4; ++nt) {
      const long col = colBase + nt * 16 + r16;
      float bv = 0.f;
      if (MODE & G_BIAS) bv = p.bias[col];
#pragma unroll
      for (int i = 0; i < 8; ++i) {
        const long row = rowBase + wv * 32 + sub * 16 + hlf * 8 + i;
        float c = acc[sub][nt][i] + bv;
        if (MODE & G_SCALE) c *= p.scale[z * p.srow + row];
        if (MODE & G_GELU)  c = 0.5f * c * (1.f + erff(c * 0.70710678f));
        const long oaddr = obase + (row / p.ordiv) * p.oro + (row % p.ordiv) * p.orow + col;
        if (MODE & G_RESID)      p.outF[oaddr] += c;
        else if (MODE & G_F32)   p.outF[oaddr]  = c;
        else                     p.outB[oaddr]  = (bf16)c;
      }
    }
  }
}

// ---------------------------------------------------------------------------
// Elementwise / reduction kernels
// ---------------------------------------------------------------------------
__global__ void k_add_pe(const float* __restrict__ xin, float* __restrict__ xout, long total) {
  long i = (long)blockIdx.x * blockDim.x + threadIdx.x;
  if (i >= total) return;
  int d  = (int)(i & (Dm - 1));
  int n  = (int)((i >> 9) & (Nn - 1));
  int k2 = d & ~1;
  float div = expf((float)k2 * (-9.21034037f / 512.f));   // 10000^(-2k/d)
  float a = (float)n * div;
  float pe = (d & 1) ? cosf(a) : sinf(a);
  xout[i] = xin[i] + pe;
}

__global__ void k_convert(const float* __restrict__ a, bf16* __restrict__ o, long n) {
  long i = (long)blockIdx.x * blockDim.x + threadIdx.x;
  if (i < n) o[i] = (bf16)a[i];
}

// proj [M, Dh] fp32 -> projT [Dh, M] bf16, folding dn = Dh^-0.25
__global__ void k_conv_projT(const float* __restrict__ proj, bf16* __restrict__ o) {
  int i = blockIdx.x * 256 + threadIdx.x;        // 64*256 total
  int d = i >> 8, m = i & 255;
  o[(long)d * Mm + m] = (bf16)(proj[(long)m * Dh + d] * 0.35355339f);
}

__global__ __launch_bounds__(128)
void k_layernorm(const float* __restrict__ x, const float* __restrict__ g,
                 const float* __restrict__ bt, bf16* __restrict__ out) {
  const int wv = threadIdx.x >> 5, lane = threadIdx.x & 31;
  const long row = (long)blockIdx.x * 4 + wv;
  const float* xr = x + row * Dm;
  float v[16]; float s = 0.f;
#pragma unroll
  for (int j = 0; j < 16; ++j) { v[j] = xr[lane + j * 32]; s += v[j]; }
#pragma unroll
  for (int o = 16; o; o >>= 1) s += __shfl_xor(s, o, 32);
  const float mu = s * (1.f / 512.f);
  float ss = 0.f;
#pragma unroll
  for (int j = 0; j < 16; ++j) { float d = v[j] - mu; ss += d * d; }
#pragma unroll
  for (int o = 16; o; o >>= 1) ss += __shfl_xor(ss, o, 32);
  const float rs = rsqrtf(ss * (1.f / 512.f) + 1e-5f);
  bf16* orow = out + row * Dm;
#pragma unroll
  for (int j = 0; j < 16; ++j) {
    int c = lane + j * 32;
    orow[c] = (bf16)((v[j] - mu) * rs * g[c] + bt[c]);
  }
}

// per (b,h,n) row: rmax over U[row,0:256], diag = 0.5*dn^2*||q_row||^2
__global__ __launch_bounds__(128)
void k_rowstats(const float* __restrict__ U, const bf16* __restrict__ qk,
                float* __restrict__ diag, float* __restrict__ rmax) {
  const int wv = threadIdx.x >> 5, lane = threadIdx.x & 31;
  const long row = (long)blockIdx.x * 4 + wv;
  const int b = (int)(row >> 14);
  const int h = (int)(row >> 11) & 7;
  const int n = (int)(row & 2047);
  const float* u = U + row * Mm;
  float m = -3.0e38f;
#pragma unroll
  for (int j = 0; j < 8; ++j) m = fmaxf(m, u[lane + j * 32]);
#pragma unroll
  for (int o = 16; o; o >>= 1) m = fmaxf(m, __shfl_xor(m, o, 32));
  const bf16* q = qk + ((long)b * Nn + n) * Dm + (long)h * Dh;
  float ss = 0.f;
#pragma unroll
  for (int j = 0; j < 2; ++j) { float v = (float)q[lane + j * 32]; ss += v * v; }
#pragma unroll
  for (int o = 16; o; o >>= 1) ss += __shfl_xor(ss, o, 32);
  if (lane == 0) { diag[row] = 0.0625f * ss; rmax[row] = m; }  // 0.5 * (1/8)
}

__global__ __launch_bounds__(256)
void k_gmax(const float* __restrict__ rmax, float* __restrict__ gmax) {
  __shared__ float red[8];
  const int z = blockIdx.x;
  float m = -3.0e38f;
#pragma unroll
  for (int j = 0; j < 8; ++j) m = fmaxf(m, rmax[(long)z * Nn + threadIdx.x + j * 256]);
#pragma unroll
  for (int o = 16; o; o >>= 1) m = fmaxf(m, __shfl_xor(m, o, 32));
  if ((threadIdx.x & 31) == 0) red[threadIdx.x >> 5] = m;
  __syncthreads();
  if (threadIdx.x == 0) {
    float g = red[0];
#pragma unroll
    for (int i = 1; i < 8; ++i) g = fmaxf(g, red[i]);
    gmax[z] = g;
  }
}

__global__ void k_feat_q(const float* __restrict__ U, const float* __restrict__ diag,
                         const float* __restrict__ rmax, bf16* __restrict__ qp, long total) {
  long i = (long)blockIdx.x * blockDim.x + threadIdx.x;
  if (i >= total) return;
  long row = i >> 8;
  float v = 0.0625f * (expf(U[i] - diag[row] - rmax[row]) + 1e-4f);  // M^-0.5 = 1/16
  qp[i] = (bf16)v;
}

// featurize k + transpose: U[b,h,n,m] -> kpT[b,h,m,n] (bf16), via LDS tile
__global__ __launch_bounds__(256)
void k_feat_kT(const float* __restrict__ U, const float* __restrict__ diag,
               const float* __restrict__ gmax, bf16* __restrict__ kpT) {
  __shared__ bf16 t[64][66];
  const int z  = blockIdx.z;
  const int n0 = blockIdx.y * 64;
  const int m0 = blockIdx.x * 64;
  const int c  = threadIdx.x & 63;
  const int r0 = threadIdx.x >> 6;   // 0..3
  const float gm = gmax[z];
#pragma unroll
  for (int it = 0; it < 16; ++it) {
    int r = it * 4 + r0;
    long row = (long)z * Nn + n0 + r;
    float u = U[row * Mm + m0 + c];
    t[r][c] = (bf16)(0.0625f * (expf(u - diag[row] - gm) + 1e-4f));
  }
  __syncthreads();
#pragma unroll
  for (int it = 0; it < 16; ++it) {
    int mr = it * 4 + r0;
    kpT[(long)z * (Mm * Nn) + (long)(m0 + mr) * Nn + n0 + c] = t[c][mr];
  }
}

__global__ __launch_bounds__(128)
void k_ksum(const bf16* __restrict__ kpT, float* __restrict__ ksum) {
  const int wv = threadIdx.x >> 5, lane = threadIdx.x & 31;
  const long row = (long)blockIdx.x * 4 + wv;       // (b,h,m)
  const bf16* r = kpT + row * Nn;
  float s = 0.f;
#pragma unroll
  for (int j = 0; j < 64; ++j) s += (float)r[lane + j * 32];
#pragma unroll
  for (int o = 16; o; o >>= 1) s += __shfl_xor(s, o, 32);
  if (lane == 0) ksum[row] = s;
}

__global__ __launch_bounds__(128)
void k_dinv(const bf16* __restrict__ qp, const float* __restrict__ ksum,
            float* __restrict__ dinv) {
  const int wv = threadIdx.x >> 5, lane = threadIdx.x & 31;
  const long row = (long)blockIdx.x * 4 + wv;       // (b,h,n)
  const bf16* q = qp + row * Mm;
  const float* ks = ksum + (row >> 11) * Mm;
  float s = 0.f;
#pragma unroll
  for (int j = 0; j < 8; ++j) s += (float)q[lane + j * 32] * ks[lane + j * 32];
#pragma unroll
  for (int o = 16; o; o >>= 1) s += __shfl_xor(s, o, 32);
  if (lane == 0) dinv[row] = 1.f / s;
}

// ---------------------------------------------------------------------------
// Host orchestration
// ---------------------------------------------------------------------------
extern "C" void kernel_launch(void* const* d_in, const int* in_sizes, int n_in,
                              void* d_out, int out_size, void* d_ws, size_t ws_size,
                              hipStream_t stream) {
  (void)in_sizes; (void)n_in; (void)out_size; (void)ws_size;
  const float* x_in = (const float*)d_in[0];
  const float* proj = (const float*)d_in[1];
  const float* ln1g = (const float*)d_in[2];
  const float* ln1b = (const float*)d_in[3];
  const float* wq   = (const float*)d_in[4];
  const float* wk   = (const float*)d_in[5];
  const float* wv_  = (const float*)d_in[6];
  const float* wo   = (const float*)d_in[7];
  const float* bo   = (const float*)d_in[8];
  const float* ln2g = (const float*)d_in[9];
  const float* ln2b = (const float*)d_in[10];
  const float* w1   = (const float*)d_in[11];
  const float* b1   = (const float*)d_in[12];
  const float* w2   = (const float*)d_in[13];
  const float* b2   = (const float*)d_in[14];
  float* x = (float*)d_out;   // residual stream lives in d_out

  // ---- workspace carve-up ----
  char* base = (char*)d_ws; size_t off = 0;
  auto alloc = [&](size_t bytes) -> void* {
    void* p = base + off; off += (bytes + 255) & ~(size_t)255; return p;
  };
  bf16* wq_bf = (bf16*)alloc(Lnum * Dm * Dm * 2);
  bf16* wk_bf = (bf16*)alloc(Lnum * Dm * Dm * 2);
  bf16* wv_bf = (bf16*)alloc(Lnum * Dm * Dm * 2);
  bf16* wo_bf = (bf16*)alloc(Lnum * Dm * Dm * 2);
  bf16* w1_bf = (bf16*)alloc(Lnum * Dm * 4 * Dm * 2);
  bf16* w2_bf = (bf16*)alloc(Lnum * 4 * Dm * Dm * 2);
  bf16* pT_bf = (bf16*)alloc(Lnum * Dh * Mm * 2);
  bf16* hbf   = (bf16*)alloc(Rr * Dm * 2);
  bf16* qbf   = (bf16*)alloc(Rr * Dm * 2);
  bf16* kbf   = (bf16*)alloc(Rr * Dm * 2);
  bf16* vbf   = (bf16*)alloc(Rr * Dm * 2);
  float* U    = (float*)alloc(BHN * Mm * 4);
  bf16* qp    = (bf16*)alloc(BHN * Mm * 2);
  bf16* kpT   = (bf16*)alloc(BH * Mm * Nn * 2);
  float* diag = (float*)alloc(BHN * 4);
  float* rmax = (float*)alloc(BHN * 4);
  float* gmax = (float*)alloc(BH * 4);
  float* ksum = (float*)alloc(BH * Mm * 4);
  float* dinv = (float*)alloc(BHN * 4);
  bf16* ctx   = (bf16*)alloc(BH * Mm * Dh * 2);
  bf16* obf   = (bf16*)alloc(Rr * Dm * 2);
  bf16* ffb   = (bf16*)alloc(Rr * 4 * Dm * 2);

  auto conv = [&](const float* s, bf16* d, long n) {
    k_convert<<<dim3((unsigned)((n + 255) / 256)), dim3(256), 0, stream>>>(s, d, n);
  };
  for (int l = 0; l < Lnum; ++l) {
    conv(wq + (long)l * Dm * Dm, wq_bf + (long)l * Dm * Dm, Dm * Dm);
    conv(wk + (long)l * Dm * Dm, wk_bf + (long)l * Dm * Dm, Dm * Dm);
    conv(wv_ + (long)l * Dm * Dm, wv_bf + (long)l * Dm * Dm, Dm * Dm);
    conv(wo + (long)l * Dm * Dm, wo_bf + (long)l * Dm * Dm, Dm * Dm);
    conv(w1 + (long)l * Dm * 4 * Dm, w1_bf + (long)l * Dm * 4 * Dm, Dm * 4 * Dm);
    conv(w2 + (long)l * 4 * Dm * Dm, w2_bf + (long)l * 4 * Dm * Dm, 4 * Dm * Dm);
    k_conv_projT<<<dim3(64), dim3(256), 0, stream>>>(proj + (long)l * Mm * Dh,
                                                     pT_bf + (long)l * Dh * Mm);
  }

  k_add_pe<<<dim3((unsigned)((Rr * Dm + 255) / 256)), dim3(256), 0, stream>>>(x_in, x, Rr * Dm);

  auto mkP = [&]() { GemmP p{}; p.azdiv = p.bzdiv = p.ozdiv = 1L << 30; p.ordiv = 1L << 30; return p; };
  auto launch = [&](int mode, const GemmP& p, long Mr, long Nc, long Z) {
    dim3 g((unsigned)(Nc / BN), (unsigned)(Mr / BM), (unsigned)Z);
    dim3 blk(128);
    switch (mode) {
      case 0:                k_gemm_wmma<0><<<g, blk, 0, stream>>>(p); break;
      case G_F32:            k_gemm_wmma<G_F32><<<g, blk, 0, stream>>>(p); break;
      case G_SCALE:          k_gemm_wmma<G_SCALE><<<g, blk, 0, stream>>>(p); break;
      case G_GELU | G_BIAS:  k_gemm_wmma<G_GELU | G_BIAS><<<g, blk, 0, stream>>>(p); break;
      case G_RESID | G_BIAS: k_gemm_wmma<G_RESID | G_BIAS><<<g, blk, 0, stream>>>(p); break;
    }
  };

  for (int l = 0; l < Lnum; ++l) {
    const bf16* Wq = wq_bf + (long)l * Dm * Dm;
    const bf16* Wk = wk_bf + (long)l * Dm * Dm;
    const bf16* Wv = wv_bf + (long)l * Dm * Dm;
    const bf16* Wo = wo_bf + (long)l * Dm * Dm;
    const bf16* W1 = w1_bf + (long)l * Dm * 4 * Dm;
    const bf16* W2 = w2_bf + (long)l * 4 * Dm * Dm;
    const bf16* Pt = pT_bf + (long)l * Dh * Mm;

    // ---- LN1 -> hbf ----
    k_layernorm<<<dim3((unsigned)(Rr / 4)), dim3(128), 0, stream>>>(
        x, ln1g + (long)l * Dm, ln1b + (long)l * Dm, hbf);

    // ---- Q, K, V projections (bf16 out) ----
    {
      GemmP p = mkP();
      p.A = hbf; p.arow = Dm; p.brow = Dm; p.orow = Dm; p.K = (int)Dm;
      p.B = Wq; p.outB = qbf; launch(0, p, Rr, Dm, 1);
      p.B = Wk; p.outB = kbf; launch(0, p, Rr, Dm, 1);
      p.B = Wv; p.outB = vbf; launch(0, p, Rr, Dm, 1);
    }

    // ---- U_q = (q * dn) @ projT  -> U[b,h,n,m] fp32 ----
    GemmP pu = mkP();
    pu.arow = Dm; pu.azi = Dh;                     // A row = (b*N+n), head slice z*64
    pu.B = Pt; pu.brow = Mm;
    pu.outF = U; pu.orow = Mm; pu.ordiv = Nn; pu.oro = Hh * Nn * Mm; pu.ozi = Nn * Mm;
    pu.K = (int)Dh;
    pu.A = qbf; launch(G_F32, pu, Rr, Mm, Hh);
    k_rowstats<<<dim3((unsigned)(BHN / 4)), dim3(128), 0, stream>>>(U, qbf, diag, rmax);
    k_feat_q<<<dim3((unsigned)((BHN * Mm + 255) / 256)), dim3(256), 0, stream>>>(
        U, diag, rmax, qp, BHN * Mm);

    // ---- U_k, k features (global-stabilized), transposed kp ----
    pu.A = kbf; launch(G_F32, pu, Rr, Mm, Hh);
    k_rowstats<<<dim3((unsigned)(BHN / 4)), dim3(128), 0, stream>>>(U, kbf, diag, rmax);
    k_gmax<<<dim3((unsigned)BH), dim3(256), 0, stream>>>(rmax, gmax);
    k_feat_kT<<<dim3(4, 32, (unsigned)BH), dim3(256), 0, stream>>>(U, diag, gmax, kpT);
    k_ksum<<<dim3((unsigned)(BH * Mm / 4)), dim3(128), 0, stream>>>(kpT, ksum);
    k_dinv<<<dim3((unsigned)(BHN / 4)), dim3(128), 0, stream>>>(qp, ksum, dinv);

    // ---- ctx[b,h,m,d] = kpT @ v ----
    {
      GemmP p = mkP();
      p.A = kpT; p.arow = Nn; p.azi = Mm * Nn;
      p.B = vbf; p.brow = Dm; p.bzdiv = Hh; p.bzo = Nn * Dm; p.bzi = Dh;
      p.outB = ctx; p.orow = Dh; p.ozi = Mm * Dh;
      p.K = (int)Nn;
      launch(0, p, Mm, Dh, BH);
    }
    // ---- o[b,n,h*64+d] = (qp @ ctx) * d_inv ----
    {
      GemmP p = mkP();
      p.A = qp; p.arow = Mm; p.azi = Nn * Mm;
      p.B = ctx; p.brow = Dh; p.bzi = Mm * Dh;
      p.outB = obf; p.orow = Dm; p.ozdiv = Hh; p.ozo = Nn * Dm; p.ozi = Dh;
      p.scale = dinv; p.srow = Nn;
      p.K = (int)Mm;
      launch(G_SCALE, p, Nn, Dh, BH);
    }
    // ---- x += o @ Wo + bo ----
    {
      GemmP p = mkP();
      p.A = obf; p.arow = Dm; p.B = Wo; p.brow = Dm;
      p.outF = x; p.orow = Dm; p.bias = bo + (long)l * Dm; p.K = (int)Dm;
      launch(G_RESID | G_BIAS, p, Rr, Dm, 1);
    }

    // ---- LN2 -> hbf ----
    k_layernorm<<<dim3((unsigned)(Rr / 4)), dim3(128), 0, stream>>>(
        x, ln2g + (long)l * Dm, ln2b + (long)l * Dm, hbf);
    // ---- FF1: gelu(h @ W1 + b1) -> ffb ----
    {
      GemmP p = mkP();
      p.A = hbf; p.arow = Dm; p.B = W1; p.brow = 4 * Dm;
      p.outB = ffb; p.orow = 4 * Dm; p.bias = b1 + (long)l * 4 * Dm; p.K = (int)Dm;
      launch(G_GELU | G_BIAS, p, Rr, 4 * Dm, 1);
    }
    // ---- FF2: x += ffb @ W2 + b2 ----
    {
      GemmP p = mkP();
      p.A = ffb; p.arow = 4 * Dm; p.B = W2; p.brow = Dm;
      p.outF = x; p.orow = Dm; p.bias = b2 + (long)l * Dm; p.K = (int)(4 * Dm);
      launch(G_RESID | G_BIAS, p, Rr, Dm, 1);
    }
  }
}
